// SpatialFormer_3444563771432
// MI455X (gfx1250) — matrix operand order repaired and checked by reading, not compiled, not run on hardware
//
#include <hip/hip_runtime.h>
#include <hip/hip_bf16.h>
#include <math.h>

// ---------------------------------------------------------------------------
// ViT block for MI455X (gfx1250, wave32, WMMA). bf16 inputs / f32 accumulate.
// Async global->LDS staging (ASYNCcnt) + double-buffered GEMM tiles.
// ---------------------------------------------------------------------------

typedef __attribute__((ext_vector_type(16))) __bf16 bf16x16;
typedef __attribute__((ext_vector_type(8)))  float  floatx8;
typedef __attribute__((ext_vector_type(4)))  int    i32x4;

#ifndef __has_builtin
#define __has_builtin(x) 0
#endif

#if __has_builtin(__builtin_amdgcn_global_load_async_to_lds_b128)
#define HAVE_ASYNC_LDS 1
#else
#define HAVE_ASYNC_LDS 0
#endif

#if HAVE_ASYNC_LDS
// 16B HBM -> LDS, no VGPR round-trip, tracked by ASYNCcnt
__device__ __forceinline__ void async_copy16(const void* g, void* l) {
    __builtin_amdgcn_global_load_async_to_lds_b128(
        (__attribute__((address_space(1))) i32x4*)(g),
        (__attribute__((address_space(3))) i32x4*)(l),
        /*offset=*/0, /*cpol=*/0);
}
#endif

__device__ __forceinline__ void wait_async0() {
#if __has_builtin(__builtin_amdgcn_s_wait_asynccnt)
    __builtin_amdgcn_s_wait_asynccnt(0);
#else
    asm volatile("s_wait_asynccnt 0x0" ::: "memory");
#endif
}

static constexpr int EPI_F32_BIAS     = 0;
static constexpr int EPI_F32_BIAS_RES = 1;
static constexpr int EPI_BF16         = 2;
static constexpr int EPI_BF16_GELU    = 3;

__device__ __forceinline__ floatx8 wmma_bf16(bf16x16 a, bf16x16 b, floatx8 c) {
    return __builtin_amdgcn_wmma_f32_16x16x32_bf16(
        /*neg_a=*/false, a, /*neg_b=*/false, b,
        /*c_mod=*/(short)0, c, /*reuse_a=*/false, /*reuse_b=*/false);
}

// A fragment (16x32, row-major source, row stride ld):
// lanes 0-15 hold K=0..15, lanes 16-31 hold K=16..31
__device__ __forceinline__ bf16x16 frag_a(const __bf16* p0, int ld, int lane) {
    int m  = lane & 15;
    int kh = (lane >> 4) * 16;
    const __bf16* p = p0 + m * ld;
    bf16x16 f;
#pragma unroll
    for (int i = 0; i < 16; i++) f[i] = p[kh + i];
    return f;
}

// B fragment (32x16) from K-contiguous storage (column n at p0 + n*ld):
// lanes 0-15 hold K=0..15, lanes 16-31 hold K=16..31
__device__ __forceinline__ bf16x16 frag_b_t(const __bf16* p0, int ld, int lane) {
    int n  = lane & 15;
    int kh = (lane >> 4) * 16;
    const __bf16* p = p0 + n * ld + kh;
    bf16x16 f;
#pragma unroll
    for (int i = 0; i < 16; i++) f[i] = p[i];
    return f;
}

// B fragment (32x16) from row-major [K][N] storage with row stride ld
__device__ __forceinline__ bf16x16 frag_b_rm(const __bf16* p0, int ld, int lane) {
    int n  = lane & 15;
    int kh = (lane >> 4) * 16;
    const __bf16* p = p0 + kh * ld + n;
    bf16x16 f;
#pragma unroll
    for (int i = 0; i < 16; i++) f[i] = p[i * ld];
    return f;
}

__device__ __forceinline__ float rmax16(float x) {
#pragma unroll
    for (int m = 1; m < 16; m <<= 1) x = fmaxf(x, __shfl_xor(x, m, 32));
    return x;
}
__device__ __forceinline__ float rsum16(float x) {
#pragma unroll
    for (int m = 1; m < 16; m <<= 1) x += __shfl_xor(x, m, 32);
    return x;
}

// ---------------------------------------------------------------------------
// Generic bf16 WMMA GEMM: C[M,N] = A[M,K] @ B[K,N] (+bias)(+res)(gelu)
// Block tile 128x128, BK=32, 8 waves (2x4), wave tile 64x32 (4x2 wmma tiles).
// M%128==0, N%128==0, K%32==0 hold for every call site.
// Async path: double-buffered LDS tiles; next tile streams in via the async
// engine while WMMAs consume the current one; one barrier per K step.
// ---------------------------------------------------------------------------
template <int EPI>
__global__ __launch_bounds__(256) void gemm_bf16_kernel(
    const __bf16* __restrict__ A, const __bf16* __restrict__ Bm,
    const float* __restrict__ bias, const float* __restrict__ res,
    float* __restrict__ outf, __bf16* __restrict__ outb,
    int M, int N, int K)
{
    const int tid  = threadIdx.x;
    const int lane = tid & 31;
    const int w    = tid >> 5;
    const int wm   = w >> 2;        // 0..1
    const int wn   = w & 3;         // 0..3
    const int m0   = blockIdx.y * 128;
    const int n0   = blockIdx.x * 128;

    floatx8 acc[4][2];
#pragma unroll
    for (int i = 0; i < 4; i++)
#pragma unroll
        for (int j = 0; j < 2; j++) acc[i][j] = {};

#if HAVE_ASYNC_LDS
    // double-buffered: A row-major [128][40], B row-major [32][136]
    __shared__ __align__(16) __bf16 As[2][128 * 40];
    __shared__ __align__(16) __bf16 Bs[2][32 * 136];

    auto issue_tile = [&](int k0, int buf) {
#pragma unroll
        for (int c = tid; c < 512; c += 256) {          // A: 128x32
            int r = c >> 2, q = c & 3;
            async_copy16(A + (size_t)(m0 + r) * K + k0 + q * 8,
                         &As[buf][r * 40 + q * 8]);
        }
#pragma unroll
        for (int c = tid; c < 512; c += 256) {          // B: 32x128
            int kk = c >> 4, nq = c & 15;
            async_copy16(Bm + (size_t)(k0 + kk) * N + n0 + nq * 8,
                         &Bs[buf][kk * 136 + nq * 8]);
        }
    };

    issue_tile(0, 0);
    int buf = 0;
    for (int k0 = 0; k0 < K; k0 += 32) {
        wait_async0();          // current tile landed in LDS
        __syncthreads();        // visible to all waves; prev compute finished
        if (k0 + 32 < K) issue_tile(k0 + 32, buf ^ 1);

        bf16x16 af[4], bfr[2];
#pragma unroll
        for (int mi = 0; mi < 4; mi++)
            af[mi] = frag_a(&As[buf][(wm * 64 + mi * 16) * 40], 40, lane);
#pragma unroll
        for (int ni = 0; ni < 2; ni++)
            bfr[ni] = frag_b_rm(&Bs[buf][wn * 32 + ni * 16], 136, lane);
#pragma unroll
        for (int mi = 0; mi < 4; mi++)
#pragma unroll
            for (int ni = 0; ni < 2; ni++)
                acc[mi][ni] = wmma_bf16(af[mi], bfr[ni], acc[mi][ni]);
        buf ^= 1;
    }
#else
    // synchronous fallback: A [128][40] row-major, B transposed [n][k] stride 40
    __shared__ __align__(16) __bf16 As1[128 * 40];
    __shared__ __align__(16) __bf16 Bs1[128 * 40];

    for (int k0 = 0; k0 < K; k0 += 32) {
#pragma unroll
        for (int c = tid; c < 512; c += 256) {
            int r = c >> 2, q = c & 3;
            *(uint4*)(&As1[r * 40 + q * 8]) =
                *(const uint4*)(A + (size_t)(m0 + r) * K + k0 + q * 8);
        }
#pragma unroll
        for (int c = tid; c < 512; c += 256) {
            int kk = c >> 4, nq = c & 15;
            __align__(16) __bf16 tmp[8];
            *(uint4*)tmp = *(const uint4*)(Bm + (size_t)(k0 + kk) * N + n0 + nq * 8);
#pragma unroll
            for (int i = 0; i < 8; i++) Bs1[(nq * 8 + i) * 40 + kk] = tmp[i];
        }
        __syncthreads();
        bf16x16 af[4], bfr[2];
#pragma unroll
        for (int mi = 0; mi < 4; mi++)
            af[mi] = frag_a(&As1[(wm * 64 + mi * 16) * 40], 40, lane);
#pragma unroll
        for (int ni = 0; ni < 2; ni++)
            bfr[ni] = frag_b_t(&Bs1[(wn * 32 + ni * 16) * 40], 40, lane);
#pragma unroll
        for (int mi = 0; mi < 4; mi++)
#pragma unroll
            for (int ni = 0; ni < 2; ni++)
                acc[mi][ni] = wmma_bf16(af[mi], bfr[ni], acc[mi][ni]);
        __syncthreads();
    }
#endif

    // Epilogue. C/D layout: VGPR v -> row v + 8*(lane>>4); col = lane&15.
    const int half = lane >> 4;
    const int colb = lane & 15;
#pragma unroll
    for (int mi = 0; mi < 4; mi++) {
#pragma unroll
        for (int ni = 0; ni < 2; ni++) {
            int rb = m0 + wm * 64 + mi * 16 + half * 8;
            int cg = n0 + wn * 32 + ni * 16 + colb;
            float bv = bias ? bias[cg] : 0.0f;
#pragma unroll
            for (int v = 0; v < 8; v++) {
                float x = acc[mi][ni][v] + bv;
                size_t idx = (size_t)(rb + v) * N + cg;
                if (EPI == EPI_F32_BIAS) {
                    outf[idx] = x;
                } else if (EPI == EPI_F32_BIAS_RES) {
                    outf[idx] = x + res[idx];
                } else if (EPI == EPI_BF16) {
                    outb[idx] = (__bf16)x;
                } else {  // exact GELU: 0.5*x*(1+erf(x/sqrt(2)))
                    float g = 0.5f * x * (1.0f + erff(x * 0.70710678118654752f));
                    outb[idx] = (__bf16)g;
                }
            }
        }
    }
}

// ---------------------------------------------------------------------------
// Attention: one workgroup per (qblock of 128 rows, head, batch).
// Full K/V (1024x64 bf16 each = 256KB) async-staged into LDS; flash-style
// online softmax per wave over its 16 query rows, WMMA for QK^T and P·V.
// ---------------------------------------------------------------------------
__global__ __launch_bounds__(256) void attn_kernel(
    const __bf16* __restrict__ qkv, __bf16* __restrict__ o)
{
    extern __shared__ __bf16 smem[];
    __bf16* Ks = smem;                    // 1024*64
    __bf16* Vs = smem + 1024 * 64;        // 1024*64
    __bf16* Qs = smem + 2 * 1024 * 64;    // 128*64
    __bf16* Ps = Qs + 128 * 64;           // 8 waves * 16*32

    const int tid  = threadIdx.x;
    const int lane = tid & 31;
    const int w    = tid >> 5;
    const int qb   = blockIdx.x;   // 0..7
    const int head = blockIdx.y;   // 0..11
    const int b    = blockIdx.z;   // 0..15
    const size_t rowbase = (size_t)b * 1024;

#if HAVE_ASYNC_LDS
    for (int c = tid; c < 8192; c += 256) {
        int n = c >> 3, q = c & 7;
        const __bf16* src = qkv + (rowbase + n) * 2304 + head * 64 + q * 8;
        async_copy16(src + 768,  &Ks[n * 64 + q * 8]);
        async_copy16(src + 1536, &Vs[n * 64 + q * 8]);
    }
    for (int c = tid; c < 1024; c += 256) {
        int r = c >> 3, q = c & 7;
        async_copy16(qkv + (rowbase + qb * 128 + r) * 2304 + head * 64 + q * 8,
                     &Qs[r * 64 + q * 8]);
    }
    wait_async0();
#else
    for (int c = tid; c < 8192; c += 256) {
        int n = c >> 3, q = c & 7;
        const __bf16* src = qkv + (rowbase + n) * 2304 + head * 64 + q * 8;
        *(uint4*)(&Ks[n * 64 + q * 8]) = *(const uint4*)(src + 768);
        *(uint4*)(&Vs[n * 64 + q * 8]) = *(const uint4*)(src + 1536);
    }
    for (int c = tid; c < 1024; c += 256) {
        int r = c >> 3, q = c & 7;
        *(uint4*)(&Qs[r * 64 + q * 8]) =
            *(const uint4*)(qkv + (rowbase + qb * 128 + r) * 2304 + head * 64 + q * 8);
    }
#endif
    __syncthreads();

    bf16x16 aq0 = frag_a(&Qs[w * 16 * 64], 64, lane);        // dims 0..31
    bf16x16 aq1 = frag_a(&Qs[w * 16 * 64 + 32], 64, lane);   // dims 32..63
    __bf16* Pw = Ps + w * 512;
    const int half = lane >> 4;
    const int colb = lane & 15;

    float ms[8], ls[8];
    floatx8 oacc[4];
#pragma unroll
    for (int v = 0; v < 8; v++) { ms[v] = -1e30f; ls[v] = 0.0f; }
#pragma unroll
    for (int c = 0; c < 4; c++) oacc[c] = {};

    const float sc = 0.125f;  // 1/sqrt(64)
    for (int j0 = 0; j0 < 1024; j0 += 32) {
        floatx8 s0 = {}, s1 = {};
        s0 = wmma_bf16(aq0, frag_b_t(&Ks[(size_t)j0 * 64], 64, lane), s0);
        s0 = wmma_bf16(aq1, frag_b_t(&Ks[(size_t)j0 * 64 + 32], 64, lane), s0);
        s1 = wmma_bf16(aq0, frag_b_t(&Ks[(size_t)(j0 + 16) * 64], 64, lane), s1);
        s1 = wmma_bf16(aq1, frag_b_t(&Ks[(size_t)(j0 + 16) * 64 + 32], 64, lane), s1);

#pragma unroll
        for (int v = 0; v < 8; v++) {
            float a0 = s0[v] * sc, a1 = s1[v] * sc;
            float mt = rmax16(fmaxf(a0, a1));
            float mn = fmaxf(ms[v], mt);
            float f  = __expf(ms[v] - mn);
            float p0 = __expf(a0 - mn);
            float p1 = __expf(a1 - mn);
            ls[v] = ls[v] * f + rsum16(p0 + p1);
            ms[v] = mn;
            Pw[(v + half * 8) * 32 + colb]      = (__bf16)p0;
            Pw[(v + half * 8) * 32 + 16 + colb] = (__bf16)p1;
#pragma unroll
            for (int c = 0; c < 4; c++) oacc[c][v] *= f;
        }
        // P (16x32) as A fragment; V rows j0..j0+31 as B fragments
        bf16x16 pf = frag_a(Pw, 32, lane);
#pragma unroll
        for (int c = 0; c < 4; c++)
            oacc[c] = wmma_bf16(pf, frag_b_rm(&Vs[(size_t)j0 * 64 + c * 16], 64, lane),
                                oacc[c]);
    }

#pragma unroll
    for (int v = 0; v < 8; v++) {
        float inv = 1.0f / ls[v];
        size_t r = rowbase + qb * 128 + w * 16 + v + half * 8;
#pragma unroll
        for (int c = 0; c < 4; c++)
            o[r * 768 + head * 64 + c * 16 + colb] = (__bf16)(oacc[c][v] * inv);
    }
}

// ---------------------------------------------------------------------------
// LayerNorm (per-row, DIM=768) -> bf16 output
// ---------------------------------------------------------------------------
__global__ __launch_bounds__(256) void ln_bf16_kernel(
    const float* __restrict__ x, const float* __restrict__ g,
    const float* __restrict__ beta, __bf16* __restrict__ y)
{
    __shared__ float rs[256], rq[256];
    const int tid = threadIdx.x;
    const size_t row = blockIdx.x;
    float s = 0.f, q = 0.f;
    for (int i = tid; i < 768; i += 256) {
        float v = x[row * 768 + i];
        s += v; q += v * v;
    }
    rs[tid] = s; rq[tid] = q;
    __syncthreads();
    for (int off = 128; off > 0; off >>= 1) {
        if (tid < off) { rs[tid] += rs[tid + off]; rq[tid] += rq[tid + off]; }
        __syncthreads();
    }
    float mean = rs[0] * (1.0f / 768.0f);
    float var  = rq[0] * (1.0f / 768.0f) - mean * mean;
    float rstd = rsqrtf(var + 1e-5f);
    for (int i = tid; i < 768; i += 256)
        y[row * 768 + i] = (__bf16)((x[row * 768 + i] - mean) * rstd * g[i] + beta[i]);
}

// ---------------------------------------------------------------------------
// Patchify: [16,3,512,512] f32 -> [16*1024, 768] bf16 (p1 p2 c ordering)
// ---------------------------------------------------------------------------
__global__ __launch_bounds__(256) void patchify_kernel(
    const float* __restrict__ x, __bf16* __restrict__ xp)
{
    size_t idx = (size_t)blockIdx.x * 256 + threadIdx.x;   // < 16384*768
    int row = (int)(idx / 768);
    int col = (int)(idx % 768);
    int b = row >> 10, pos = row & 1023;
    int hp = pos >> 5, wp = pos & 31;
    int p1 = col / 48, rem = col - p1 * 48;
    int p2 = rem / 3,  c   = rem - p2 * 3;
    size_t src = (((size_t)(b * 3 + c) * 512) + hp * 16 + p1) * 512 + wp * 16 + p2;
    xp[idx] = (__bf16)x[src];
}

__global__ __launch_bounds__(256) void f32_to_bf16_kernel(
    const float* __restrict__ a, __bf16* __restrict__ b, int n)
{
    int i = blockIdx.x * 256 + threadIdx.x;
    if (i < n) b[i] = (__bf16)a[i];
}

// ---------------------------------------------------------------------------
extern "C" void kernel_launch(void* const* d_in, const int* in_sizes, int n_in,
                              void* d_out, int out_size, void* d_ws, size_t ws_size,
                              hipStream_t stream)
{
    (void)in_sizes; (void)n_in; (void)out_size; (void)ws_size;
    const float* x       = (const float*)d_in[0];
    const float* w_patch = (const float*)d_in[1];
    const float* b_patch = (const float*)d_in[2];
    const float* g1      = (const float*)d_in[3];
    const float* beta1   = (const float*)d_in[4];
    const float* w_qkv   = (const float*)d_in[5];
    const float* w_out   = (const float*)d_in[6];
    const float* b_out   = (const float*)d_in[7];
    const float* g2      = (const float*)d_in[8];
    const float* beta2   = (const float*)d_in[9];
    const float* w_ff1   = (const float*)d_in[10];
    const float* b_ff1   = (const float*)d_in[11];
    const float* w_ff2   = (const float*)d_in[12];
    const float* b_ff2   = (const float*)d_in[13];
    float* out = (float*)d_out;

    const int M = 16384;   // B * N_patches = 16 * 1024
    char* ws = (char*)d_ws;
    size_t off = 0;
    auto alloc = [&](size_t bytes) -> void* {
        void* p = ws + off;
        off += (bytes + 255) & ~(size_t)255;
        return p;
    };
    __bf16* xpat  = (__bf16*)alloc((size_t)M * 768 * 2);
    float*  t     = (float*) alloc((size_t)M * 768 * 4);
    __bf16* lnb   = (__bf16*)alloc((size_t)M * 768 * 2);
    __bf16* qkvb  = (__bf16*)alloc((size_t)M * 2304 * 2);
    __bf16* attno = (__bf16*)alloc((size_t)M * 768 * 2);
    __bf16* ffa   = (__bf16*)alloc((size_t)M * 3072 * 2);
    __bf16* wpb   = (__bf16*)alloc((size_t)768 * 768 * 2);
    __bf16* wqkvb = (__bf16*)alloc((size_t)768 * 2304 * 2);
    __bf16* woutb = (__bf16*)alloc((size_t)768 * 768 * 2);
    __bf16* wff1b = (__bf16*)alloc((size_t)768 * 3072 * 2);
    __bf16* wff2b = (__bf16*)alloc((size_t)3072 * 768 * 2);

    auto cvt = [&](const float* a, __bf16* b, int n) {
        f32_to_bf16_kernel<<<(n + 255) / 256, 256, 0, stream>>>(a, b, n);
    };
    cvt(w_patch, wpb,   768 * 768);
    cvt(w_qkv,   wqkvb, 768 * 2304);
    cvt(w_out,   woutb, 768 * 768);
    cvt(w_ff1,   wff1b, 768 * 3072);
    cvt(w_ff2,   wff2b, 3072 * 768);

    // 1. patchify + patch-embed GEMM -> t
    patchify_kernel<<<(M * 768) / 256, 256, 0, stream>>>(x, xpat);
    gemm_bf16_kernel<EPI_F32_BIAS><<<dim3(768 / 128, M / 128), 256, 0, stream>>>(
        xpat, wpb, b_patch, nullptr, t, nullptr, M, 768, 768);

    // 2. LN1 -> bf16, QKV GEMM -> bf16
    ln_bf16_kernel<<<M, 256, 0, stream>>>(t, g1, beta1, lnb);
    gemm_bf16_kernel<EPI_BF16><<<dim3(2304 / 128, M / 128), 256, 0, stream>>>(
        lnb, wqkvb, nullptr, nullptr, nullptr, qkvb, M, 2304, 768);

    // 3. attention (K+V+Q+P scratch = 286720 B dynamic LDS, fits 320KB WGP LDS)
    const int attn_lds = (2 * 1024 * 64 + 128 * 64 + 8 * 512) * 2;
    (void)hipFuncSetAttribute((const void*)attn_kernel,
                              hipFuncAttributeMaxDynamicSharedMemorySize, attn_lds);
    attn_kernel<<<dim3(8, 12, 16), 256, attn_lds, stream>>>(qkvb, attno);

    // 4. out projection + residual -> d_out (t2)
    gemm_bf16_kernel<EPI_F32_BIAS_RES><<<dim3(768 / 128, M / 128), 256, 0, stream>>>(
        attno, woutb, b_out, t, out, nullptr, M, 768, 768);

    // 5. LN2 -> bf16, FF1 + GELU -> bf16
    ln_bf16_kernel<<<M, 256, 0, stream>>>(out, g2, beta2, lnb);
    gemm_bf16_kernel<EPI_BF16_GELU><<<dim3(3072 / 128, M / 128), 256, 0, stream>>>(
        lnb, wff1b, b_ff1, nullptr, nullptr, ffa, M, 3072, 768);

    // 6. FF2 + residual (in-place on d_out: each element read once, then written)
    gemm_bf16_kernel<EPI_F32_BIAS_RES><<<dim3(768 / 128, M / 128), 256, 0, stream>>>(
        ffa, wff2b, b_ff2, out, out, nullptr, M, 768, 3072);
}
